// MegatronMLP_69337952026974
// MI455X (gfx1250) — matrix operand order repaired and checked by reading, not compiled, not run on hardware
//
#include <hip/hip_runtime.h>
#include <hip/hip_bf16.h>
#include <math.h>

#define HIDDEN 1024
#define FFN    4096
#define NE     8
#define TOPK   2
#define S_TOK  16384          // 8 * 2048
#define CAP    4096           // TOPK * S_TOK / NE
#define LDST   40             // LDS row stride in bf16 elements (80B, 16B aligned, conflict-skewed)
#define LBUF   (128 * LDST)   // one LDS tile buffer (bf16 elements)

typedef __attribute__((ext_vector_type(16))) __bf16 v16bf;
typedef __attribute__((ext_vector_type(8)))  __bf16 v8bf;
typedef __attribute__((ext_vector_type(4)))  __bf16 v4bf;
typedef __attribute__((ext_vector_type(8)))  float  v8f;

__device__ inline __bf16 f2bf(float f) {
  union { float f; unsigned u; } in; in.f = f;
  unsigned u = in.u;
  u += 0x7FFFu + ((u >> 16) & 1u);   // round to nearest even
  union { unsigned short s; __bf16 b; } out;
  out.s = (unsigned short)(u >> 16);
  return out.b;
}

// ---------------------------------------------------------------- zero output
__global__ void moe_zero_f32(float* p, int n) {
  int i = blockIdx.x * 256 + threadIdx.x;
  if (i < n) p[i] = 0.0f;
}

// ---------------------------------------------------------------- gating: one wave per token
__global__ void __launch_bounds__(256) moe_gate(const float* __restrict__ x,
                                                const float* __restrict__ wg,
                                                int* __restrict__ topi,
                                                float* __restrict__ topw) {
  const int lane = threadIdx.x & 31;
  const int wave = threadIdx.x >> 5;
  const int s = blockIdx.x * 8 + wave;
  float p[NE];
#pragma unroll
  for (int e = 0; e < NE; ++e) p[e] = 0.0f;
  const float* xr = x + (size_t)s * HIDDEN;
  for (int i = lane; i < HIDDEN; i += 32) {
    float xv = xr[i];
    const float* wr = wg + i * NE;
#pragma unroll
    for (int e = 0; e < NE; ++e) p[e] += xv * wr[e];
  }
#pragma unroll
  for (int off = 16; off > 0; off >>= 1)
#pragma unroll
    for (int e = 0; e < NE; ++e) p[e] += __shfl_xor(p[e], off, 32);
  if (lane == 0) {
    int i0 = 0; float v0 = p[0];
#pragma unroll
    for (int e = 1; e < NE; ++e) if (p[e] > v0) { v0 = p[e]; i0 = e; }
    int i1 = -1; float v1 = -1e30f;
#pragma unroll
    for (int e = 0; e < NE; ++e) if (e != i0 && p[e] > v1) { v1 = p[e]; i1 = e; }
    float w0 = 1.0f / (1.0f + __expf(v1 - v0));   // softmax over top-2, renormalized
    topi[s * 2 + 0] = i0;  topi[s * 2 + 1] = i1;
    topw[s * 2 + 0] = w0;  topw[s * 2 + 1] = 1.0f - w0;
  }
}

// ---------------------------------------------------------------- routing scan: one wave per expert
__global__ void __launch_bounds__(256) moe_route(const int* __restrict__ topi,
                                                 const float* __restrict__ topw,
                                                 int* __restrict__ slot_token,
                                                 float* __restrict__ slot_weight) {
  for (int i = threadIdx.x; i < NE * CAP; i += 256) {
    slot_token[i] = -1;
    slot_weight[i] = 0.0f;
  }
  __syncthreads();
  const int e = threadIdx.x >> 5;
  const int lane = threadIdx.x & 31;
  int base = 0;
  // reference ordering: all k=0 assignments first, then k=1 (offs = cumsum over K)
  for (int k = 0; k < TOPK; ++k) {
    for (int s0 = 0; s0 < S_TOK; s0 += 32) {
      int s = s0 + lane;
      int hit = (topi[s * 2 + k] == e) ? 1 : 0;
      unsigned m = (unsigned)__ballot(hit);
      int pos = base + __popc(m & ((1u << lane) - 1u));
      if (hit && pos < CAP) {
        slot_token[e * CAP + pos] = s;
        slot_weight[e * CAP + pos] = topw[s * 2 + k];
      }
      base += __popc(m);
    }
  }
}

// ---------------------------------------------------------------- gather tokens -> bf16 expert buffer
__global__ void __launch_bounds__(256) moe_gather(const float* __restrict__ x,
                                                  const int* __restrict__ slot_token_e,
                                                  __bf16* __restrict__ xbuf) {
  const int c = blockIdx.x;
  const int t = slot_token_e[c];
  const int i = threadIdx.x * 4;
  v4bf o;
  if (t >= 0) {
    float4 v = *(const float4*)&x[(size_t)t * HIDDEN + i];
    o[0] = f2bf(v.x); o[1] = f2bf(v.y); o[2] = f2bf(v.z); o[3] = f2bf(v.w);
  } else {
    o[0] = o[1] = o[2] = o[3] = f2bf(0.0f);
  }
  *(v4bf*)&xbuf[(size_t)c * HIDDEN + i] = o;
}

// ---------------------------------------------------------------- W (K x N fp32) -> Wt (N x K bf16)
__global__ void __launch_bounds__(256) moe_wconv(const float* __restrict__ W,
                                                 __bf16* __restrict__ Wt,
                                                 int K, int N) {
  size_t tid = (size_t)blockIdx.x * 256 + threadIdx.x;   // N*K/8 threads
  int kb = (int)(tid / (size_t)N);
  int n  = (int)(tid % (size_t)N);
  v8bf o;
#pragma unroll
  for (int i = 0; i < 8; ++i) o[i] = f2bf(W[(size_t)(kb * 8 + i) * N + n]);
  *(v8bf*)&Wt[(size_t)n * K + kb * 8] = o;
}

// ---------------------------------------------------------------- shared WMMA mainloop (double-buffered LDS)
// A: M x K row-major (bf16). B: N x K row-major (bf16, K-contiguous = pre-transposed weights).
// Block = 256 threads = 8 waves; 128x128 tile; each wave owns 32 rows x 64 cols (2x4 WMMA tiles).
// Pipeline: global loads for slab k+1 are issued right after the barrier and parked in
// registers; they are committed to the other LDS buffer at the top of the next iteration,
// so HBM latency is hidden behind a full WMMA compute phase. One barrier per K-step.
template <int K>
__device__ inline void gemm_mainloop(const __bf16* __restrict__ A,
                                     const __bf16* __restrict__ B,
                                     int m0, int n0,
                                     __bf16* lA, __bf16* lB,   // each 2*LBUF elements
                                     v8f acc[2][4]) {
  const int tid  = threadIdx.x;
  const int lane = tid & 31;
  const int wave = tid >> 5;
  const int waveRow = wave & 3;      // 4 groups of 32 rows
  const int waveCol = wave >> 2;     // 2 groups of 64 cols
  const int kh  = lane >> 4;         // K-half select per CDNA5 bf16 fragment layout
  const int l15 = lane & 15;

  // tile load assignment: 512 chunks of 8 bf16 per operand; 2 chunks/thread (rows r0, r0+64)
  const int r0 = tid >> 2, kc = (tid & 3) * 8;
  const __bf16* pa0 = A + (size_t)(m0 + r0)      * K + kc;
  const __bf16* pa1 = A + (size_t)(m0 + r0 + 64) * K + kc;
  const __bf16* pb0 = B + (size_t)(n0 + r0)      * K + kc;
  const __bf16* pb1 = B + (size_t)(n0 + r0 + 64) * K + kc;

  // prologue: stage slab 0
  v8bf sa0 = *(const v8bf*)pa0;
  v8bf sa1 = *(const v8bf*)pa1;
  v8bf sb0 = *(const v8bf*)pb0;
  v8bf sb1 = *(const v8bf*)pb1;

#pragma unroll 2
  for (int k0 = 0; k0 < K; k0 += 32) {
    const int buf = (k0 >> 5) & 1;
    __bf16* cA = lA + buf * LBUF;
    __bf16* cB = lB + buf * LBUF;

    // commit staged slab to this iteration's LDS buffer
    *(v8bf*)&cA[r0 * LDST + kc]        = sa0;
    *(v8bf*)&cA[(r0 + 64) * LDST + kc] = sa1;
    *(v8bf*)&cB[r0 * LDST + kc]        = sb0;
    *(v8bf*)&cB[(r0 + 64) * LDST + kc] = sb1;
    __syncthreads();

    // kick off global loads for the next slab (land before next iteration's stores)
    if (k0 + 32 < K) {
      sa0 = *(const v8bf*)(pa0 + k0 + 32);
      sa1 = *(const v8bf*)(pa1 + k0 + 32);
      sb0 = *(const v8bf*)(pb0 + k0 + 32);
      sb1 = *(const v8bf*)(pb1 + k0 + 32);
    }

    // fragments + 8 WMMAs from current buffer
    v16bf af[2], bf4[4];
#pragma unroll
    for (int mt = 0; mt < 2; ++mt) {
      int m = waveRow * 32 + mt * 16 + l15;            // lane holds row m
      v8bf lo = *(v8bf*)&cA[m * LDST + kh * 8];        // K = kh*8 .. +7
      v8bf hi = *(v8bf*)&cA[m * LDST + 16 + kh * 8];   // K = 16+kh*8 .. +7
#pragma unroll
      for (int i = 0; i < 8; ++i) { af[mt][i] = lo[i]; af[mt][i + 8] = hi[i]; }
    }
#pragma unroll
    for (int nt = 0; nt < 4; ++nt) {
      int n = waveCol * 64 + nt * 16 + l15;            // lane holds column n
      v8bf lo = *(v8bf*)&cB[n * LDST + kh * 8];        // K = kh*8 .. +7
      v8bf hi = *(v8bf*)&cB[n * LDST + 16 + kh * 8];   // K = 16+kh*8 .. +7
#pragma unroll
      for (int i = 0; i < 8; ++i) { bf4[nt][i] = lo[i]; bf4[nt][i + 8] = hi[i]; }
    }
#pragma unroll
    for (int mt = 0; mt < 2; ++mt)
#pragma unroll
      for (int nt = 0; nt < 4; ++nt)
        acc[mt][nt] = __builtin_amdgcn_wmma_f32_16x16x32_bf16(
            false, af[mt], false, bf4[nt], (short)0, acc[mt][nt], false, false);
  }
}

// ---------------------------------------------------------------- GEMM1: h = GELU(xbuf @ W1 + b1), bf16 out
__global__ void __launch_bounds__(256) moe_gemm1(const __bf16* __restrict__ A,
                                                 const __bf16* __restrict__ Bt,
                                                 const float* __restrict__ b1e,
                                                 __bf16* __restrict__ h) {
  __shared__ __align__(16) __bf16 lA[2 * LBUF];
  __shared__ __align__(16) __bf16 lB[2 * LBUF];
  const int m0 = blockIdx.y * 128;
  const int n0 = blockIdx.x * 128;
  v8f acc[2][4] = {};
  gemm_mainloop<HIDDEN>(A, Bt, m0, n0, lA, lB, acc);

  const int lane = threadIdx.x & 31;
  const int wave = threadIdx.x >> 5;
  const int waveRow = wave & 3, waveCol = wave >> 2;
#pragma unroll
  for (int mt = 0; mt < 2; ++mt)
#pragma unroll
    for (int nt = 0; nt < 4; ++nt) {
      int col = n0 + waveCol * 64 + nt * 16 + (lane & 15);
      float bv = b1e[col];
#pragma unroll
      for (int i = 0; i < 8; ++i) {
        int row = m0 + waveRow * 32 + mt * 16 + (lane >> 4) * 8 + i;
        float v = acc[mt][nt][i] + bv;
        v = 0.5f * v * (1.0f + erff(v * 0.70710678118654752f));   // exact GELU
        h[(size_t)row * FFN + col] = f2bf(v);
      }
    }
}

// ---------------------------------------------------------------- GEMM2: out[token] += w * (h @ W2 + b2)
__global__ void __launch_bounds__(256) moe_gemm2(const __bf16* __restrict__ A,
                                                 const __bf16* __restrict__ Bt,
                                                 const float* __restrict__ b2e,
                                                 const int* __restrict__ slot_token_e,
                                                 const float* __restrict__ slot_weight_e,
                                                 float* __restrict__ out) {
  __shared__ __align__(16) __bf16 lA[2 * LBUF];
  __shared__ __align__(16) __bf16 lB[2 * LBUF];
  const int m0 = blockIdx.y * 128;
  const int n0 = blockIdx.x * 128;
  v8f acc[2][4] = {};
  gemm_mainloop<FFN>(A, Bt, m0, n0, lA, lB, acc);

  const int lane = threadIdx.x & 31;
  const int wave = threadIdx.x >> 5;
  const int waveRow = wave & 3, waveCol = wave >> 2;
#pragma unroll
  for (int mt = 0; mt < 2; ++mt)
#pragma unroll
    for (int nt = 0; nt < 4; ++nt) {
      int col = n0 + waveCol * 64 + nt * 16 + (lane & 15);
      float bv = b2e[col];
#pragma unroll
      for (int i = 0; i < 8; ++i) {
        int row = m0 + waveRow * 32 + mt * 16 + (lane >> 4) * 8 + i;
        int t = slot_token_e[row];
        if (t >= 0) {
          float w = slot_weight_e[row];
          out[(size_t)t * HIDDEN + col] += w * (acc[mt][nt][i] + bv);
        }
      }
    }
}

// ---------------------------------------------------------------- host-side orchestration
extern "C" void kernel_launch(void* const* d_in, const int* in_sizes, int n_in,
                              void* d_out, int out_size, void* d_ws, size_t ws_size,
                              hipStream_t stream) {
  const float* x  = (const float*)d_in[0];   // (8,2048,1024)
  const float* Wg = (const float*)d_in[1];   // (1024,8)
  const float* W1 = (const float*)d_in[2];   // (8,1024,4096)
  const float* b1 = (const float*)d_in[3];   // (8,4096)
  const float* W2 = (const float*)d_in[4];   // (8,4096,1024)
  const float* b2 = (const float*)d_in[5];   // (8,1024)
  float* out = (float*)d_out;                // (8,2048,1024) + 1024 zeros

  // workspace layout (~49 MB)
  char* w = (char*)d_ws;
  int*    topi        = (int*)w;          w += (size_t)2 * S_TOK * 4;
  float*  topw        = (float*)w;        w += (size_t)2 * S_TOK * 4;
  int*    slot_token  = (int*)w;          w += (size_t)NE * CAP * 4;
  float*  slot_weight = (float*)w;        w += (size_t)NE * CAP * 4;
  __bf16* xbuf        = (__bf16*)w;       w += (size_t)CAP * HIDDEN * 2;
  __bf16* wbuf        = (__bf16*)w;       w += (size_t)HIDDEN * FFN * 2;   // reused W1t / W2t
  __bf16* hbuf        = (__bf16*)w;       // CAP * FFN * 2

  moe_zero_f32<<<(out_size + 255) / 256, 256, 0, stream>>>(out, out_size);
  moe_gate<<<S_TOK / 8, 256, 0, stream>>>(x, Wg, topi, topw);
  moe_route<<<1, 256, 0, stream>>>(topi, topw, slot_token, slot_weight);

  const int wconv_blocks = (HIDDEN * FFN / 8) / 256;   // 2048
  for (int e = 0; e < NE; ++e) {
    const int* ste = slot_token + (size_t)e * CAP;
    const float* swe = slot_weight + (size_t)e * CAP;

    moe_gather<<<CAP, 256, 0, stream>>>(x, ste, xbuf);

    // W1[e]: (d x f), transpose-convert -> (f x d) bf16
    moe_wconv<<<wconv_blocks, 256, 0, stream>>>(W1 + (size_t)e * HIDDEN * FFN, wbuf, HIDDEN, FFN);
    moe_gemm1<<<dim3(FFN / 128, CAP / 128), 256, 0, stream>>>(xbuf, wbuf, b1 + (size_t)e * FFN, hbuf);

    // W2[e]: (f x d), transpose-convert -> (d x f) bf16
    moe_wconv<<<wconv_blocks, 256, 0, stream>>>(W2 + (size_t)e * FFN * HIDDEN, wbuf, FFN, HIDDEN);
    moe_gemm2<<<dim3(HIDDEN / 128, CAP / 128), 256, 0, stream>>>(hbuf, wbuf, b2 + (size_t)e * HIDDEN,
                                                                 ste, swe, out);
  }
}
